// ShiftedWindows_MSA_2D_67499706024013
// MI455X (gfx1250) — matrix operand order repaired and checked by reading, not compiled, not run on hardware
//
#include <hip/hip_runtime.h>
#include <hip/hip_bf16.h>
#include <math.h>

#define WSZ   7
#define SSZ   3
#define NH    4
#define CDIM  128
#define HD    32
#define NWIN  8      // windows per side
#define HDIM  56
#define NPAD  64
#define NTOK  49

typedef __attribute__((ext_vector_type(16))) __bf16 v16bf;
typedef __attribute__((ext_vector_type(8)))  float  v8f;

union BF16Frag { v16bf v; unsigned short u[16]; uint4 q[2]; };

__device__ __forceinline__ unsigned short f2bf(float f) {
    __bf16 h = (__bf16)f;                       // native cvt (RNE)
    return __builtin_bit_cast(unsigned short, h);
}

// A-matrix 16x32 bf16 fragment from row-major LDS (lane = M; lane half h holds
// contiguous K = 16h .. 16h+15, matching the B loaders). Two 16B ds_load_b128.
__device__ __forceinline__ v16bf load_a_lds(const unsigned short* p, int ld, int m0, int k0, int lane) {
    const int m = lane & 15, half = lane >> 4;
    const unsigned short* rp = p + (m0 + m) * ld + k0 + 16 * half;
    BF16Frag f;
    f.q[0] = *(const uint4*)(rp);        // K = k0 + 16h .. +7
    f.q[1] = *(const uint4*)(rp + 8);    // K = k0 + 16h + 8 .. +15
    return f.v;
}

// B[k][n] = src[(n0+n)*ld + k0 + k]  (transposed source, e.g. K^T / Vt): lane = N.
// 16 contiguous ushorts per lane -> two 16B loads.
__device__ __forceinline__ v16bf load_bT_lds(const unsigned short* p, int ld, int n0, int k0, int lane) {
    const int n = lane & 15, half = lane >> 4;
    const unsigned short* rp = p + (n0 + n) * ld + k0 + 16 * half;
    BF16Frag f;
    f.q[0] = *(const uint4*)(rp);
    f.q[1] = *(const uint4*)(rp + 8);
    return f.v;
}

// Weight B-fragment from PRE-CONVERTED bf16 global W[j][c]; B[k=c][n=j].
// 16 contiguous ushorts -> two global_load_b128, no conversion VALU.
__device__ __forceinline__ v16bf load_bW_bf(const unsigned short* __restrict__ W, int ldw,
                                            int j0, int c0, int lane) {
    const int n = lane & 15, half = lane >> 4;
    const unsigned short* rp = W + (j0 + n) * ldw + c0 + 16 * half;
    BF16Frag f;
    f.q[0] = *(const uint4*)(rp);
    f.q[1] = *(const uint4*)(rp + 8);
    return f.v;
}

// ---- pre-pass: convert fp32 weights -> bf16 workspace (once per launch) ----
__global__ void cvt_weights(const float* __restrict__ wqkv, const float* __restrict__ wproj,
                            unsigned short* __restrict__ wsq, unsigned short* __restrict__ wsp) {
    int i = blockIdx.x * 256 + threadIdx.x;
    if (i < 3 * CDIM * CDIM) wsq[i] = f2bf(wqkv[i]);
    if (i < CDIM * CDIM)     wsp[i] = f2bf(wproj[i]);
}

__global__ __launch_bounds__(256, 2)
void swin_msa_fused(const float* __restrict__ x, const float* __restrict__ rpb,
                    const unsigned short* __restrict__ wqkv_bf,
                    const unsigned short* __restrict__ wproj_bf,
                    const float* __restrict__ bproj, float* __restrict__ out)
{
    extern __shared__ __align__(16) char smem[];
    float*          sS  = (float*)smem;                                // [NH][64][64] f32  (64 KB)
    unsigned short* sX  = (unsigned short*)(smem + NH * 64 * 64 * 4);  // [64][128] bf16    (16 KB) -> reused for O
    unsigned short* sQ  = sX + 64 * CDIM;                              // [64][128] bf16    (16 KB)
    unsigned short* sK  = sQ + 64 * CDIM;                              // [64][128] bf16    (16 KB)
    unsigned short* sVt = sK + 64 * CDIM;                              // [128][64] bf16 V^T(16 KB)
    unsigned short* sP  = sVt + CDIM * 64;                             // [NH][64][64] bf16 (32 KB)

    const int tid  = threadIdx.x;
    const int lane = tid & 31, wv = tid >> 5;
    const int bid  = blockIdx.x;
    const int wi   = bid & 63, b = bid >> 6;
    const int wh   = wi >> 3, ww = wi & 7;

    // ---- stage 0: load shifted window (roll -3 baked in), fp32 -> bf16, pad rows 49..63 ----
    for (int i = tid; i < NPAD * CDIM; i += 256) {
        int m = i >> 7, c = i & (CDIM - 1);
        unsigned short v = 0;
        if (m < NTOK) {
            int lr = m / WSZ, lc = m % WSZ;
            int hh = (wh * WSZ + lr + SSZ) % HDIM;
            int wc = (ww * WSZ + lc + SSZ) % HDIM;
            v = f2bf(x[(((long)b * HDIM + hh) * HDIM + wc) * CDIM + c]);
        }
        sX[i] = v;
    }
    __syncthreads();

    const float scale = 0.17677669529663689f;   // hd^-0.5

    // ---- stage 1: QKV = X(64x128) @ Wqkv^T ; Q pre-scaled; V stored transposed ----
    for (int t = wv; t < 96; t += 8) {
        int mt = t & 3, jt = t >> 2, j0 = jt * 16;
        v8f acc = {0, 0, 0, 0, 0, 0, 0, 0};
#pragma unroll
        for (int kk = 0; kk < 4; ++kk) {
            v16bf a  = load_a_lds(sX, CDIM, mt * 16, kk * 32, lane);
            v16bf bb = load_bW_bf(wqkv_bf, CDIM, j0, kk * 32, lane);
            acc = __builtin_amdgcn_wmma_f32_16x16x32_bf16(false, a, false, bb, (short)0, acc, false, false);
        }
        int n = lane & 15, half = lane >> 4;
        if (j0 < CDIM) {                     // Q (scaled)
#pragma unroll
            for (int r = 0; r < 8; ++r) {
                int m = mt * 16 + r + 8 * half;
                sQ[m * CDIM + j0 + n] = f2bf(acc[r] * scale);
            }
        } else if (j0 < 2 * CDIM) {          // K
#pragma unroll
            for (int r = 0; r < 8; ++r) {
                int m = mt * 16 + r + 8 * half;
                sK[m * CDIM + (j0 - CDIM) + n] = f2bf(acc[r]);
            }
        } else {                             // V -> transposed, packed 16B store
            union { uint4 q; unsigned short u[8]; } pk;
#pragma unroll
            for (int r = 0; r < 8; ++r) pk.u[r] = f2bf(acc[r]);
            int d = (j0 - 2 * CDIM) + n;     // 0..127
            *(uint4*)(sVt + d * 64 + mt * 16 + 8 * half) = pk.q;
        }
    }
    __syncthreads();

    // ---- stage 2: S = Q @ K^T per head (hd=32 -> one WMMA/tile) + bias + mask ----
    for (int t = wv; t < 64; t += 8) {
        int h = t >> 4, rt = (t >> 2) & 3, ct = t & 3;
        v16bf a  = load_a_lds(sQ, CDIM, rt * 16, h * HD, lane);
        v16bf bb = load_bT_lds(sK, CDIM, ct * 16, h * HD, lane);
        v8f acc = {0, 0, 0, 0, 0, 0, 0, 0};
        acc = __builtin_amdgcn_wmma_f32_16x16x32_bf16(false, a, false, bb, (short)0, acc, false, false);
        int n = ct * 16 + (lane & 15), half = lane >> 4;
#pragma unroll
        for (int r = 0; r < 8; ++r) {
            int m = rt * 16 + r + 8 * half;
            float val = -3.0e38f;
            if (m < NTOK && n < NTOK) {
                int mr = m / WSZ, mc = m % WSZ, nr = n / WSZ, nc = n % WSZ;
                int idx = (mr - nr + WSZ - 1) * (2 * WSZ - 1) + (mc - nc + WSZ - 1);
                float bias = rpb[idx * NH + h];
                int lm = ((wh < NWIN - 1) ? 0 : ((mr < WSZ - SSZ) ? 3 : 6))
                       + ((ww < NWIN - 1) ? 0 : ((mc < WSZ - SSZ) ? 1 : 2));
                int ln = ((wh < NWIN - 1) ? 0 : ((nr < WSZ - SSZ) ? 3 : 6))
                       + ((ww < NWIN - 1) ? 0 : ((nc < WSZ - SSZ) ? 1 : 2));
                val = acc[r] + bias + ((lm == ln) ? 0.0f : -100.0f);
            }
            sS[(h * 64 + m) * 64 + n] = val;
        }
    }
    __syncthreads();

    // ---- stage 3: row softmax (one thread per head-row), write P bf16 ----
    {
        int h = tid >> 6, m = tid & 63;
        const float* row = sS + (h * 64 + m) * 64;
        unsigned short* prow = sP + (h * 64 + m) * 64;
        if (m < NTOK) {
            float mx = -3.0e38f;
#pragma unroll 8
            for (int n = 0; n < 64; ++n) mx = fmaxf(mx, row[n]);
            float sum = 0.f;
#pragma unroll 8
            for (int n = 0; n < 64; ++n) sum += __expf(row[n] - mx);
            float inv = 1.0f / sum;
#pragma unroll 8
            for (int n = 0; n < 64; ++n) prow[n] = f2bf(__expf(row[n] - mx) * inv);
        } else {
#pragma unroll 8
            for (int n = 0; n < 64; ++n) prow[n] = 0;
        }
    }
    __syncthreads();

    // ---- stage 4: O = P @ V per head (K=64 -> 2 WMMA), B from transposed V ----
    for (int t = wv; t < 32; t += 8) {
        int h = t >> 3, rt = (t >> 1) & 3, dt = t & 1;
        v8f acc = {0, 0, 0, 0, 0, 0, 0, 0};
#pragma unroll
        for (int kk = 0; kk < 2; ++kk) {
            v16bf a  = load_a_lds(sP + h * 64 * 64, 64, rt * 16, kk * 32, lane);
            v16bf bb = load_bT_lds(sVt, 64, h * HD + dt * 16, kk * 32, lane);
            acc = __builtin_amdgcn_wmma_f32_16x16x32_bf16(false, a, false, bb, (short)0, acc, false, false);
        }
        int n = lane & 15, half = lane >> 4;
#pragma unroll
        for (int r = 0; r < 8; ++r) {
            int m = rt * 16 + r + 8 * half;
            sX[m * CDIM + h * HD + dt * 16 + n] = f2bf(acc[r]);
        }
    }
    __syncthreads();

    // ---- stage 5: Y = O @ Wproj^T + b, write with reverse-shift (roll +3) addressing ----
    for (int t = wv; t < 32; t += 8) {
        int mt = t >> 3, jt = t & 7;
        v8f acc = {0, 0, 0, 0, 0, 0, 0, 0};
#pragma unroll
        for (int kk = 0; kk < 4; ++kk) {
            v16bf a  = load_a_lds(sX, CDIM, mt * 16, kk * 32, lane);
            v16bf bb = load_bW_bf(wproj_bf, CDIM, jt * 16, kk * 32, lane);
            acc = __builtin_amdgcn_wmma_f32_16x16x32_bf16(false, a, false, bb, (short)0, acc, false, false);
        }
        int n = lane & 15, half = lane >> 4;
        int j = jt * 16 + n;
        float bj = bproj[j];
#pragma unroll
        for (int r = 0; r < 8; ++r) {
            int m = mt * 16 + r + 8 * half;
            if (m < NTOK) {
                int lr = m / WSZ, lc = m % WSZ;
                int hh = (wh * WSZ + lr + SSZ) % HDIM;
                int wc = (ww * WSZ + lc + SSZ) % HDIM;
                out[(((long)b * HDIM + hh) * HDIM + wc) * CDIM + j] = acc[r] + bj;
            }
        }
    }
}

extern "C" void kernel_launch(void* const* d_in, const int* in_sizes, int n_in,
                              void* d_out, int out_size, void* d_ws, size_t ws_size,
                              hipStream_t stream) {
    (void)in_sizes; (void)n_in; (void)out_size; (void)ws_size;
    const float* x     = (const float*)d_in[0];
    const float* rpb   = (const float*)d_in[1];
    const float* wqkv  = (const float*)d_in[2];
    const float* wproj = (const float*)d_in[3];
    const float* bproj = (const float*)d_in[4];
    float* out = (float*)d_out;

    unsigned short* wsq = (unsigned short*)d_ws;          // 384x128 bf16
    unsigned short* wsp = wsq + 3 * CDIM * CDIM;          // 128x128 bf16   (128 KB total)

    // 1) convert weights to bf16 (L2-resident for all 2048 window blocks)
    cvt_weights<<<(3 * CDIM * CDIM + 255) / 256, 256, 0, stream>>>(wqkv, wproj, wsq, wsp);

    // 2) fused shifted-window MSA, one block per window
    const int numBlocks = 32 * NWIN * NWIN;               // B * nW = 2048
    const size_t smemBytes = (size_t)(NH * 64 * 64 * 4)   // sS
                           + (size_t)(64 * CDIM * 2) * 4  // sX, sQ, sK, sVt
                           + (size_t)(NH * 64 * 64 * 2);  // sP   = 160 KB
    swin_msa_fused<<<numBlocks, 256, smemBytes, stream>>>(x, rpb, wsq, wsp, bproj, out);
}
